// HybridAttention_32195074850900
// MI455X (gfx1250) — matrix operand order repaired and checked by reading, not compiled, not run on hardware
//
#include <hip/hip_runtime.h>
#include <hip/hip_bf16.h>
#include <math.h>

#define BB 4
#define TT 2048
#define CC 768
#define HH 12
#define DD 64
#define WINSZ 256
#define MTOT (BB*TT)   // 8192

typedef __bf16 bf16_t;
typedef __attribute__((ext_vector_type(16))) __bf16 v16bf;
typedef __attribute__((ext_vector_type(8)))  __bf16 v8bf;
typedef __attribute__((ext_vector_type(8)))  float  v8f;
typedef __attribute__((ext_vector_type(4)))  float  f32x4;
typedef __attribute__((ext_vector_type(4)))  unsigned int u32x4;
typedef __attribute__((ext_vector_type(8)))  int    i32x8;
typedef __attribute__((ext_vector_type(4)))  int    i32x4;

__device__ __forceinline__ v8f zero8() {
  v8f z = {0.f,0.f,0.f,0.f,0.f,0.f,0.f,0.f};
  return z;
}

__device__ __forceinline__ v8f wmma_bf16(v16bf a, v16bf b, v8f c) {
  return __builtin_amdgcn_wmma_f32_16x16x32_bf16(
      false, a, false, b, (short)0, c, false, false);
}

// ---- Fragment loaders (contiguous k order k = 16*(lane/16) + s, wave32) ----
// A fragment (16x32 bf16): per lane two contiguous 16B runs -> 2x b128 loads.
__device__ __forceinline__ v16bf a_frag_bf16(const bf16_t* A, int row0, int k0, int ld) {
  int lane = threadIdx.x & 31;
  const bf16_t* p = A + (size_t)(row0 + (lane & 15)) * ld + k0 + ((lane >> 4) << 4);
  v8bf lo = *(const v8bf*)p;
  v8bf hi = *(const v8bf*)(p + 8);
  return __builtin_shufflevector(lo, hi, 0,1,2,3,4,5,6,7,8,9,10,11,12,13,14,15);
}

// A fragment from f32 source: 4x b128 loads + cvt.
__device__ __forceinline__ v16bf a_frag_f32(const float* A, int row0, int k0, int ld) {
  int lane = threadIdx.x & 31;
  const float* p = A + (size_t)(row0 + (lane & 15)) * ld + k0 + ((lane >> 4) << 4);
  f32x4 a0 = *(const f32x4*)p;
  f32x4 a1 = *(const f32x4*)(p + 4);
  f32x4 b0 = *(const f32x4*)(p + 8);
  f32x4 b1 = *(const f32x4*)(p + 12);
  v16bf a;
#pragma unroll
  for (int e = 0; e < 4; ++e) {
    a[e]      = (bf16_t)a0[e];
    a[4 + e]  = (bf16_t)a1[e];
    a[8 + e]  = (bf16_t)b0[e];
    a[12 + e] = (bf16_t)b1[e];
  }
  return a;
}

// B fragment from fragment-major packed storage: 2x b128 loads.
// Packed layout: tile-index * 512 + lane*16 + e  (e in WMMA operand order).
__device__ __forceinline__ v16bf b_frag_packed(const bf16_t* P, int tile) {
  int lane = threadIdx.x & 31;
  const v8bf* p = (const v8bf*)(P + (size_t)tile * 512 + lane * 16);
  return __builtin_shufflevector(p[0], p[1], 0,1,2,3,4,5,6,7,8,9,10,11,12,13,14,15);
}

// B fragment of K^T (element (kd,j) = K[j][kd]): 16 contiguous f32 per lane.
__device__ __forceinline__ v16bf bT_frag_f32_guard(const float* Kp, int kd0, int j0,
                                                   int ld, int jmax) {
  int lane = threadIdx.x & 31;
  int j    = j0 + (lane & 15);
  int khi  = (lane >> 4) << 4;
  bool ok  = (j < jmax);
  const float* p = Kp + (size_t)(ok ? j : 0) * ld + kd0 + khi;
  v16bf b;
  if (ok) {
#pragma unroll
    for (int q4 = 0; q4 < 4; ++q4) {
      f32x4 t = *(const f32x4*)(p + 4 * q4);
#pragma unroll
      for (int e = 0; e < 4; ++e) b[4 * q4 + e] = (bf16_t)t[e];
    }
  } else {
#pragma unroll
    for (int e = 0; e < 16; ++e) b[e] = (bf16_t)0.0f;
  }
  return b;
}

// B fragment (32x16) from f32 row-major source, row-guarded (strided; V tiles).
__device__ __forceinline__ v16bf b_frag_f32_rowguard(const float* Vm, int k0, int col0,
                                                     int ld, int kmax) {
  int lane = threadIdx.x & 31;
  int col  = col0 + (lane & 15);
  int khi  = (lane >> 4) << 4;
  v16bf b;
#pragma unroll
  for (int r = 0; r < 8; ++r) {
    int kk = k0 + khi + 2 * r;
    b[2*r]   = (kk     < kmax) ? (bf16_t)Vm[(size_t)kk * ld + col]       : (bf16_t)0.0f;
    b[2*r+1] = (kk + 1 < kmax) ? (bf16_t)Vm[(size_t)(kk + 1) * ld + col] : (bf16_t)0.0f;
  }
  return b;
}

// ---- TDM: async 2-D tile load (rows x cols f32) global -> LDS ----
// Descriptor per CDNA5 ISA 8.3-8.6: count=1, type=2, data_size=4B.
__device__ __forceinline__ void tdm_load_2d_f32(unsigned lds_addr, const float* gptr,
                                                unsigned cols, unsigned rows,
                                                unsigned row_stride_elems) {
  unsigned long long ga = (unsigned long long)(size_t)gptr;
  u32x4 g0;
  g0[0] = 1u;                                   // count=1, user mode
  g0[1] = lds_addr;                             // LDS byte address
  g0[2] = (unsigned)(ga & 0xffffffffu);         // global_addr[31:0]
  g0[3] = (unsigned)((ga >> 32) & 0x01ffffffu) | (2u << 30);  // addr[56:32] | type=2
  i32x8 g1;
  g1[0] = (int)(2u << 16);                      // data_size = 4 bytes
  g1[1] = (int)(cols << 16);                    // tensor_dim0[15:0]
  g1[2] = (int)(rows << 16);                    // tensor_dim1[15:0]
  g1[3] = (int)(cols << 16);                    // tile_dim0
  g1[4] = (int)rows;                            // tile_dim1 (tile_dim2 = 0)
  g1[5] = (int)row_stride_elems;                // tensor_dim0_stride[31:0]
  g1[6] = 0;
  g1[7] = 0;
  i32x4 z4 = {0, 0, 0, 0};
#if defined(__clang_major__) && __clang_major__ >= 23
  i32x8 z8 = {0, 0, 0, 0, 0, 0, 0, 0};
  __builtin_amdgcn_tensor_load_to_lds(g0, g1, z4, z4, z8, 0);
#else
  __builtin_amdgcn_tensor_load_to_lds(g0, g1, z4, z4, 0);
#endif
}

// ---------------- Kernel: f32 -> bf16 conversion ----------------
__global__ void k_cvt_bf16(const float* __restrict__ in, bf16_t* __restrict__ out, int n) {
  int i = blockIdx.x * blockDim.x + threadIdx.x;
  if (i < n) out[i] = (bf16_t)in[i];
}

// ---------------- Kernel: repack f32 weights -> fragment-major bf16 ----------------
// One thread per (ktile, ntile, lane); pays the strided reads once.
__global__ void k_pack_b(const float* __restrict__ W, bf16_t* __restrict__ P,
                         int K, int N) {
  int idx = blockIdx.x * blockDim.x + threadIdx.x;
  int total = (K / 32) * (N / 16) * 32;
  if (idx >= total) return;
  int lane = idx & 31;
  int tile = idx >> 5;
  int nt = tile % (N / 16);
  int kt = tile / (N / 16);
  int col = nt * 16 + (lane & 15);
  int khi = (lane >> 4) << 4;
  bf16_t* dst = P + (size_t)tile * 512 + lane * 16;
#pragma unroll
  for (int r = 0; r < 8; ++r) {
    int kk = kt * 32 + khi + 2 * r;
    dst[2*r]   = (bf16_t)W[(size_t)kk * N + col];
    dst[2*r+1] = (bf16_t)W[(size_t)(kk + 1) * N + col];
  }
}

// ---------------- Kernel: QKV GEMM (bf16 WMMA, packed B) ----------------
__global__ void __launch_bounds__(128)
k_qkv_wmma(const bf16_t* __restrict__ X, const bf16_t* __restrict__ Wp,
           const float* __restrict__ bias,
           float* __restrict__ q, float* __restrict__ k, float* __restrict__ v) {
  int wave = threadIdx.x >> 5, lane = threadIdx.x & 31;
  int m0 = blockIdx.y * 64 + wave * 16;
  int n0 = blockIdx.x * 64;
  const int NT = 3 * CC / 16;   // B tiles per K-slab
  v8f acc[4] = {zero8(), zero8(), zero8(), zero8()};
  for (int k0 = 0; k0 < CC; k0 += 32) {
    v16bf a = a_frag_bf16(X, m0, k0, CC);
    if (k0 + 32 < CC)
      __builtin_prefetch(X + (size_t)(m0 + (lane & 15)) * CC + k0 + 32, 0, 1);
    int tbase = (k0 >> 5) * NT + (n0 >> 4);
#pragma unroll
    for (int nt = 0; nt < 4; ++nt)
      acc[nt] = wmma_bf16(a, b_frag_packed(Wp, tbase + nt), acc[nt]);
  }
  int hi8 = (lane >> 4) << 3;
#pragma unroll
  for (int nt = 0; nt < 4; ++nt) {
    int n = n0 + nt * 16 + (lane & 15);
    float bv = bias[n];
    int which = n / CC;
    int cc = n - which * CC;
    int h = cc >> 6, d = cc & 63;
    float* dst = (which == 0) ? q : (which == 1) ? k : v;
#pragma unroll
    for (int r = 0; r < 8; ++r) {
      int m = m0 + hi8 + r;
      int b_ = m >> 11, t = m & (TT - 1);
      dst[(((size_t)b_ * HH + h) * TT + t) * DD + d] = acc[nt][r] + bv;
    }
  }
}

// ---------------- Kernel: windowed attention ----------------
#define SW 296   // LDS row stride; 296*2 bytes = 37*16 keeps b128 frag loads aligned
__global__ void __launch_bounds__(32)
k_attn(const float* __restrict__ q, const float* __restrict__ k,
       const float* __restrict__ v, float* __restrict__ lout) {
  __shared__ __align__(16) float  sS[16 * SW];
  __shared__ __align__(16) bf16_t sP[16 * SW];
  __shared__ float sLinv[16];

  int bh = blockIdx.x >> 7;
  int qt = blockIdx.x & 127;
  int q0 = qt << 4;
  int lane = threadIdx.x & 31;
  int hi8 = (lane >> 4) << 3;

  const float* Q = q + (size_t)bh * TT * DD;
  const float* K = k + (size_t)bh * TT * DD;
  const float* V = v + (size_t)bh * TT * DD;

  v16bf qa0 = a_frag_f32(Q, q0, 0, DD);
  v16bf qa1 = a_frag_f32(Q, q0, 32, DD);
  const float scale = 0.125f;

  for (int p = 0; p < 9; ++p) {
#pragma unroll
    for (int sub = 0; sub < 2; ++sub) {
      int j0 = q0 + p * 32 + sub * 16;
      v8f s = zero8();
      s = wmma_bf16(qa0, bT_frag_f32_guard(K, 0, j0, DD, TT), s);
      s = wmma_bf16(qa1, bT_frag_f32_guard(K, 32, j0, DD, TT), s);
      int col = p * 32 + sub * 16 + (lane & 15);
#pragma unroll
      for (int r = 0; r < 8; ++r)
        sS[(hi8 + r) * SW + col] = s[r] * scale;
    }
  }
  __syncthreads();

  if (lane < 16) {
    int row = lane;
    int cjlo = row;
    int cjhi = row + WINSZ;
    int cjmax = TT - q0;
    if (cjhi > cjmax) cjhi = cjmax;
    float m = -1e30f;
    for (int cj = cjlo; cj < cjhi; ++cj)
      m = fmaxf(m, sS[row * SW + cj]);
    float sum = 0.f;
    for (int cj = 0; cj < 288; ++cj) {
      float pv = 0.f;
      if (cj >= cjlo && cj < cjhi) {
        pv = __expf(sS[row * SW + cj] - m);
        sum += pv;
      }
      sP[row * SW + cj] = (bf16_t)pv;
    }
    sLinv[row] = 1.0f / sum;
  }
  __syncthreads();

  v8f o[4] = {zero8(), zero8(), zero8(), zero8()};
  for (int p = 0; p < 9; ++p) {
    v16bf pa = a_frag_bf16(sP, 0, p * 32, SW);
#pragma unroll
    for (int nt = 0; nt < 4; ++nt) {
      v16bf vb = b_frag_f32_rowguard(V, q0 + p * 32, nt * 16, DD, TT);
      o[nt] = wmma_bf16(pa, vb, o[nt]);
    }
  }

  float* Outp = lout + (size_t)bh * TT * DD;
#pragma unroll
  for (int nt = 0; nt < 4; ++nt) {
    int col = nt * 16 + (lane & 15);
#pragma unroll
    for (int r = 0; r < 8; ++r) {
      int row = hi8 + r;
      Outp[(size_t)(q0 + row) * DD + col] = o[nt][r] * sLinv[row];
    }
  }
}

// ---------------- Kernel: GRU-style scan ----------------
// 48 rows over 3 workgroups; per-step k/v tiles staged by TDM (async tensor
// load to LDS), weights LDS-resident in fragment-major bf16.
__global__ void __launch_bounds__(128)
k_gru(const float* __restrict__ kg, const float* __restrict__ vg,
      const float* __restrict__ Wr, const float* __restrict__ Ur,
      const float* __restrict__ Wz, const float* __restrict__ Wn,
      const float* __restrict__ Un, float* __restrict__ rnn) {
  __shared__ __align__(16) bf16_t sWr[8*512], sUr[8*512], sWz[8*512], sWn[8*512], sUn[8*512];
  __shared__ __align__(16) bf16_t sHb[16*64], sKb[16*64], sRH[16*64];
  __shared__ __align__(16) float  sKf[16*64], sVf[16*64];

  int tid  = threadIdx.x;
  int lane = tid & 31;
  int nt   = tid >> 5;
  int rbase = blockIdx.x * 16;

  // Pack the five 64x64 weight matrices into LDS fragment-major bf16.
  for (int s = tid; s < 256; s += 128) {
    int pl = s & 31, tile = s >> 5;
    int kt = tile >> 2, ct = tile & 3;
    int col = ct * 16 + (pl & 15);
    int khi = (pl >> 4) << 4;
    int base = tile * 512 + pl * 16;
#pragma unroll
    for (int r = 0; r < 8; ++r) {
      int kk = kt * 32 + khi + 2 * r;
      sWr[base + 2*r] = (bf16_t)Wr[kk*64 + col]; sWr[base + 2*r+1] = (bf16_t)Wr[(kk+1)*64 + col];
      sUr[base + 2*r] = (bf16_t)Ur[kk*64 + col]; sUr[base + 2*r+1] = (bf16_t)Ur[(kk+1)*64 + col];
      sWz[base + 2*r] = (bf16_t)Wz[kk*64 + col]; sWz[base + 2*r+1] = (bf16_t)Wz[(kk+1)*64 + col];
      sWn[base + 2*r] = (bf16_t)Wn[kk*64 + col]; sWn[base + 2*r+1] = (bf16_t)Wn[(kk+1)*64 + col];
      sUn[base + 2*r] = (bf16_t)Un[kk*64 + col]; sUn[base + 2*r+1] = (bf16_t)Un[(kk+1)*64 + col];
    }
  }
  for (int i = tid; i < 16 * 64; i += 128) sHb[i] = (bf16_t)0.0f;

  float hreg[8];
#pragma unroll
  for (int r = 0; r < 8; ++r) hreg[r] = 0.f;

  int hi8  = (lane >> 4) << 3;
  int colg = (nt << 4) + (lane & 15);
  unsigned ldsKf = (unsigned)(size_t)(void*)sKf;
  unsigned ldsVf = (unsigned)(size_t)(void*)sVf;

  for (int t = 0; t < TT; ++t) {
    __syncthreads();   // prev-step readers of sKf/sVf/sKb done; sHb final
    if ((tid >> 5) == 0) {   // wave-uniform: only wave 0 issues TDM ops
      const float* kbase = kg + ((size_t)rbase * TT + t) * DD;
      const float* vbase = vg + ((size_t)rbase * TT + t) * DD;
      tdm_load_2d_f32(ldsKf, kbase, DD, 16, TT * DD);
      tdm_load_2d_f32(ldsVf, vbase, DD, 16, TT * DD);
      __builtin_amdgcn_s_wait_tensorcnt(0);
    }
    __syncthreads();
    for (int i = tid; i < 16 * 64; i += 128) sKb[i] = (bf16_t)sKf[i];
    __syncthreads();

    v16bf hA0 = a_frag_bf16(sHb, 0, 0, 64);
    v16bf hA1 = a_frag_bf16(sHb, 0, 32, 64);
    v16bf kA0 = a_frag_bf16(sKb, 0, 0, 64);
    v16bf kA1 = a_frag_bf16(sKb, 0, 32, 64);

    v8f rp = zero8(), zp = zero8();
    rp = wmma_bf16(hA0, b_frag_packed(sWr, 0 + nt), rp);
    rp = wmma_bf16(hA1, b_frag_packed(sWr, 4 + nt), rp);
    rp = wmma_bf16(kA0, b_frag_packed(sUr, 0 + nt), rp);
    rp = wmma_bf16(kA1, b_frag_packed(sUr, 4 + nt), rp);
    zp = wmma_bf16(hA0, b_frag_packed(sWz, 0 + nt), zp);
    zp = wmma_bf16(hA1, b_frag_packed(sWz, 4 + nt), zp);

#pragma unroll
    for (int r = 0; r < 8; ++r) {
      float rv = 1.0f / (1.0f + __expf(-rp[r]));
      sRH[(hi8 + r) * 64 + colg] = (bf16_t)(rv * hreg[r]);
    }
    __syncthreads();

    v16bf rA0 = a_frag_bf16(sRH, 0, 0, 64);
    v16bf rA1 = a_frag_bf16(sRH, 0, 32, 64);
    v8f np = zero8();
    np = wmma_bf16(rA0, b_frag_packed(sWn, 0 + nt), np);
    np = wmma_bf16(rA1, b_frag_packed(sWn, 4 + nt), np);
    np = wmma_bf16(kA0, b_frag_packed(sUn, 0 + nt), np);
    np = wmma_bf16(kA1, b_frag_packed(sUn, 4 + nt), np);

#pragma unroll
    for (int r = 0; r < 8; ++r) {
      int row = hi8 + r;
      float kvv = sKf[row * 64 + colg];
      float vvv = sVf[row * 64 + colg];
      float z  = 1.0f / (1.0f + __expf(-(zp[r] + kvv)));
      float n  = tanhf(np[r]);
      float hn = (1.0f - z) * hreg[r] + z * (n * vvv);
      hreg[r] = hn;
      sHb[row * 64 + colg] = (bf16_t)hn;
      rnn[((size_t)(rbase + row) * TT + t) * DD + colg] = hn;
    }
  }
}

// ---------------- Kernel: gate alpha ----------------
__global__ void k_alpha(const float* __restrict__ x, const float* __restrict__ gw,
                        const float* __restrict__ gb, float* __restrict__ alpha) {
  int idx = blockIdx.x * blockDim.x + threadIdx.x;
  if (idx >= MTOT * HH) return;
  int bt = idx / HH;
  int h  = idx - bt * HH;
  const float* xr = x + (size_t)bt * CC;
  float acc = gb[h];
  for (int c = 0; c < CC; ++c) acc += xr[c] * gw[c * HH + h];
  alpha[idx] = 1.0f / (1.0f + __expf(-acc));
}

// ---------------- Kernel: combine ----------------
__global__ void k_combine(const float* __restrict__ lo, const float* __restrict__ rn,
                          const float* __restrict__ alpha, bf16_t* __restrict__ y) {
  int idx = blockIdx.x * blockDim.x + threadIdx.x;
  if (idx >= MTOT * CC) return;
  int bt = idx / CC;
  int c  = idx - bt * CC;
  int h = c >> 6, d = c & 63;
  int b = bt >> 11, t = bt & (TT - 1);
  size_t gi = (((size_t)b * HH + h) * TT + t) * DD + d;
  float a = alpha[bt * HH + h];
  y[idx] = (bf16_t)(a * lo[gi] + (1.0f - a) * rn[gi]);
}

// ---------------- Kernel: output projection (bf16 WMMA, packed B) ----------------
__global__ void __launch_bounds__(128)
k_proj_wmma(const bf16_t* __restrict__ Y, const bf16_t* __restrict__ Wp,
            const float* __restrict__ bias, float* __restrict__ out) {
  int wave = threadIdx.x >> 5, lane = threadIdx.x & 31;
  int m0 = blockIdx.y * 64 + wave * 16;
  int n0 = blockIdx.x * 64;
  const int NT = CC / 16;
  v8f acc[4] = {zero8(), zero8(), zero8(), zero8()};
  for (int k0 = 0; k0 < CC; k0 += 32) {
    v16bf a = a_frag_bf16(Y, m0, k0, CC);
    if (k0 + 32 < CC)
      __builtin_prefetch(Y + (size_t)(m0 + (lane & 15)) * CC + k0 + 32, 0, 1);
    int tbase = (k0 >> 5) * NT + (n0 >> 4);
#pragma unroll
    for (int nt = 0; nt < 4; ++nt)
      acc[nt] = wmma_bf16(a, b_frag_packed(Wp, tbase + nt), acc[nt]);
  }
  int hi8 = (lane >> 4) << 3;
#pragma unroll
  for (int nt = 0; nt < 4; ++nt) {
    int n = n0 + nt * 16 + (lane & 15);
    float bv = bias[n];
#pragma unroll
    for (int r = 0; r < 8; ++r) {
      int m = m0 + hi8 + r;
      out[(size_t)m * CC + n] = acc[nt][r] + bv;
    }
  }
}

extern "C" void kernel_launch(void* const* d_in, const int* in_sizes, int n_in,
                              void* d_out, int out_size, void* d_ws, size_t ws_size,
                              hipStream_t stream) {
  const float* x      = (const float*)d_in[0];
  const float* qkv_w  = (const float*)d_in[2];
  const float* qkv_b  = (const float*)d_in[3];
  const float* proj_w = (const float*)d_in[4];
  const float* proj_b = (const float*)d_in[5];
  const float* Wr     = (const float*)d_in[6];
  const float* Ur     = (const float*)d_in[7];
  const float* Wz     = (const float*)d_in[8];
  const float* Wn     = (const float*)d_in[9];
  const float* Un     = (const float*)d_in[10];
  const float* gate_w = (const float*)d_in[11];
  const float* gate_b = (const float*)d_in[12];

  char* ws = (char*)d_ws;
  size_t off = 0;
  auto alloc = [&](size_t bytes) -> char* {
    char* p = ws + off;
    off = (off + bytes + 255) & ~(size_t)255;
    return p;
  };

  bf16_t* xb  = (bf16_t*)alloc((size_t)MTOT * CC * 2);
  bf16_t* qwp = (bf16_t*)alloc((size_t)CC * 3 * CC * 2);   // packed qkv weights
  bf16_t* pwp = (bf16_t*)alloc((size_t)CC * CC * 2);       // packed proj weights
  float*  qf  = (float*)alloc((size_t)MTOT * CC * 4);
  float*  kf  = (float*)alloc((size_t)MTOT * CC * 4);
  float*  vf  = (float*)alloc((size_t)MTOT * CC * 4);
  float*  lo  = (float*)alloc((size_t)MTOT * CC * 4);
  float*  rn  = (float*)alloc((size_t)MTOT * CC * 4);
  float*  al  = (float*)alloc((size_t)MTOT * HH * 4);
  bf16_t* yb  = (bf16_t*)alloc((size_t)MTOT * CC * 2);

  int nx = MTOT * CC;
  k_cvt_bf16<<<(nx + 255) / 256, 256, 0, stream>>>(x, xb, nx);
  int nqp = (CC / 32) * (3 * CC / 16) * 32;
  k_pack_b<<<(nqp + 255) / 256, 256, 0, stream>>>(qkv_w, qwp, CC, 3 * CC);
  int npp = (CC / 32) * (CC / 16) * 32;
  k_pack_b<<<(npp + 255) / 256, 256, 0, stream>>>(proj_w, pwp, CC, CC);

  k_qkv_wmma<<<dim3(3 * CC / 64, MTOT / 64), 128, 0, stream>>>(xb, qwp, qkv_b, qf, kf, vf);
  k_attn<<<BB * HH * (TT / 16), 32, 0, stream>>>(qf, kf, vf, lo);
  k_gru<<<3, 128, 0, stream>>>(kf, vf, Wr, Ur, Wz, Wn, Un, rn);
  k_alpha<<<(MTOT * HH + 255) / 256, 256, 0, stream>>>(x, gate_w, gate_b, al);
  k_combine<<<(MTOT * CC + 255) / 256, 256, 0, stream>>>(lo, rn, al, yb);
  k_proj_wmma<<<dim3(CC / 64, MTOT / 64), 128, 0, stream>>>(yb, pwp, proj_b, (float*)d_out);
}
